// EquivariantGNN_17300128268844
// MI455X (gfx1250) — matrix operand order repaired and checked by reading, not compiled, not run on hardware
//
#include <hip/hip_runtime.h>
#include <stdint.h>

// ---------------------------------------------------------------------------
// EGNN forward for MI455X (gfx1250): bf16 WMMA (f32 accumulate) edge/node MLPs,
// f32 atomics for segment sums, fp32 residual state. wave32 / 16x16x32 tiles.
// ---------------------------------------------------------------------------

typedef __bf16 bf16_t;
typedef bf16_t v16bf __attribute__((ext_vector_type(16)));
typedef float  v8f   __attribute__((ext_vector_type(8)));

#define DD 64
#define MSG_K 160        // 133 padded to 5*32
#define MSG_STRIDE 168   // LDS row stride (bf16 elems) to stagger banks
#define U_K 128
#define U_STRIDE 136
#define W2_STRIDE 72
#define NWAVES 4         // waves per block (128 threads)

__device__ __forceinline__ unsigned short f2bf(float f) {
  unsigned u = __builtin_bit_cast(unsigned, f);
  u += 0x7FFFu + ((u >> 16) & 1u);          // round-to-nearest-even
  return (unsigned short)(u >> 16);
}

union Frag { v16bf v; unsigned u[8]; };

// A fragment (16x32 bf16): lane holds row m=lane&15; VGPR v holds K pair
// k = 2v + 8*hi (+8 for v>=4)  [ISA 7.12.2 "16-bit A-Matrix 16x32"]
__device__ __forceinline__ void load_afrag(Frag& a, const unsigned short* row,
                                           int kb, int hi) {
#pragma unroll
  for (int v = 0; v < 8; ++v) {
    int k = 2 * v + 8 * hi + ((v >= 4) ? 8 : 0);
    a.u[v] = *(const unsigned*)(row + kb + k);
  }
}

// B fragment (32x16 bf16) from a transposed weight row WT[n][k]:
// VGPR v holds K pair k = 2v + 8*hi (+8 for v>=4), same as A.
__device__ __forceinline__ void load_bfrag(Frag& b, const unsigned short* wtRow,
                                           int kb, int hi) {
#pragma unroll
  for (int v = 0; v < 8; ++v)
    b.u[v] = *(const unsigned*)(wtRow + kb + 2 * v + 8 * hi + ((v >= 4) ? 8 : 0));
}

__device__ __forceinline__ v8f wmma_bf16(const Frag& a, const Frag& b, v8f c) {
  return __builtin_amdgcn_wmma_f32_16x16x32_bf16(false, a.v, false, b.v,
                                                 (short)0, c, false, false);
}

// ---------------------------------------------------------------------------
// h = x @ W_in + b_in
__global__ void k_embed(const float* __restrict__ x, const float* __restrict__ W,
                        const float* __restrict__ b, float* __restrict__ h, int N) {
  long idx = (long)blockIdx.x * blockDim.x + threadIdx.x;
  if (idx >= (long)N * DD) return;
  int n = (int)(idx >> 6), d = (int)(idx & 63);
  float a = b[d];
#pragma unroll
  for (int k = 0; k < 11; ++k) a += x[(long)n * 11 + k] * W[k * DD + d];
  h[idx] = a;
}

__global__ void k_prep(const float* __restrict__ pos, float* __restrict__ pos_c,
                       float* __restrict__ deg, int N) {
  int i = blockIdx.x * blockDim.x + threadIdx.x;
  if (i < N * 3) pos_c[i] = pos[i];
  if (i < N) deg[i] = 0.f;
}

__global__ void k_deg(const int* __restrict__ dst, float* __restrict__ deg, int E) {
  int e = blockIdx.x * blockDim.x + threadIdx.x;
  if (e < E) atomicAdd(&deg[dst[e]], 1.0f);
}

__global__ void k_zero(float* __restrict__ aggr, float* __restrict__ pos_a, int N) {
  long i = (long)blockIdx.x * blockDim.x + threadIdx.x;
  if (i < (long)N * DD) aggr[i] = 0.f;
  if (i < (long)N * 3) pos_a[i] = 0.f;
}

__global__ void k_out(const float* __restrict__ h, const float* __restrict__ W,
                      const float* __restrict__ b, float* __restrict__ out, int N) {
  int n = blockIdx.x * blockDim.x + threadIdx.x;
  if (n >= N) return;
  float a = b[0];
#pragma unroll
  for (int d = 0; d < DD; ++d) a += h[(long)n * DD + d] * W[d];
  out[n] = a;
}

// ---------------------------------------------------------------------------
// Edge kernel: per 16-edge tile (1 wave):
//  m = relu(relu([h_i|h_j|ea|dist] @ Wm1 + bm1) @ Wm2 + bm2)
//  atomicAdd aggr[dst] += m ;  p = m.Wp + bp ; atomicAdd pos_aggr[dst] += rel*p
__global__ void __launch_bounds__(128)
k_edge(const float* __restrict__ h, const float* __restrict__ pos,
       const float* __restrict__ ea, const int* __restrict__ src,
       const int* __restrict__ dst,
       const float* __restrict__ Wm1, const float* __restrict__ bm1,
       const float* __restrict__ Wm2, const float* __restrict__ bm2,
       const float* __restrict__ Wp, const float* __restrict__ bp,
       float* __restrict__ aggr, float* __restrict__ pos_aggr,
       int E, int msg_in) {
  __shared__ unsigned short sW1[DD][MSG_STRIDE];   // Wm1^T bf16 (zero-padded K)
  __shared__ unsigned short sW2[DD][W2_STRIDE];    // Wm2^T bf16
  __shared__ float sWp[DD], sB1[DD], sB2[DD];
  __shared__ unsigned short sM[NWAVES][16][MSG_STRIDE]; // per-wave A staging
  __shared__ float sRel[NWAVES][16][3];
  __shared__ int   sDst[NWAVES][16];

  const int tid = threadIdx.x;
  const int wave = tid >> 5, lane = tid & 31;
  const int hi = lane >> 4, lm = lane & 15;

  for (int idx = tid; idx < MSG_K * DD; idx += blockDim.x) {
    int k = idx >> 6, n = idx & 63;
    float w = (k < msg_in) ? Wm1[k * DD + n] : 0.f;
    sW1[n][k] = f2bf(w);
  }
  for (int idx = tid; idx < DD * DD; idx += blockDim.x) {
    int k = idx >> 6, n = idx & 63;
    sW2[n][k] = f2bf(Wm2[k * DD + n]);
  }
  if (tid < DD) { sWp[tid] = Wp[tid]; sB1[tid] = bm1[tid]; sB2[tid] = bm2[tid]; }
  __syncthreads();
  const float bpv = bp[0];

  const int nTiles = (E + 15) >> 4;
  for (int tile = blockIdx.x * NWAVES + wave; tile < nTiles;
       tile += gridDim.x * NWAVES) {
    const int base = tile << 4;
    { // ---- stage 16 edges into bf16 LDS tile (2 lanes per edge row) ----
      int r = lane >> 1, half = lane & 1;
      int e = base + r;
      bool valid = (e < E);
      int ec = valid ? e : (E - 1);
      int j = src[ec], i = dst[ec];
      unsigned short* mrow = &sM[wave][r][0];
      const float4* hi4 = (const float4*)(h + (long)i * DD) + half * 8;
      const float4* hj4 = (const float4*)(h + (long)j * DD) + half * 8;
#pragma unroll
      for (int c = 0; c < 8; ++c) {
        float4 f = hi4[c];
        unsigned p0 = f2bf(f.x) | ((unsigned)f2bf(f.y) << 16);
        unsigned p1 = f2bf(f.z) | ((unsigned)f2bf(f.w) << 16);
        *(uint2*)(mrow + half * 32 + 4 * c) = make_uint2(p0, p1);
        f = hj4[c];
        p0 = f2bf(f.x) | ((unsigned)f2bf(f.y) << 16);
        p1 = f2bf(f.z) | ((unsigned)f2bf(f.w) << 16);
        *(uint2*)(mrow + 64 + half * 32 + 4 * c) = make_uint2(p0, p1);
      }
      if (half == 0) {
        float rx = pos[(long)i * 3 + 0] - pos[(long)j * 3 + 0];
        float ry = pos[(long)i * 3 + 1] - pos[(long)j * 3 + 1];
        float rz = pos[(long)i * 3 + 2] - pos[(long)j * 3 + 2];
        float dist = sqrtf(rx * rx + ry * ry + rz * rz);
        mrow[128] = f2bf(ea[(long)ec * 4 + 0]);
        mrow[129] = f2bf(ea[(long)ec * 4 + 1]);
        mrow[130] = f2bf(ea[(long)ec * 4 + 2]);
        mrow[131] = f2bf(ea[(long)ec * 4 + 3]);
        mrow[132] = f2bf(dist);
#pragma unroll
        for (int c = 133; c < MSG_K; ++c) mrow[c] = 0;
        sRel[wave][r][0] = rx; sRel[wave][r][1] = ry; sRel[wave][r][2] = rz;
        sDst[wave][r] = valid ? i : -1;
      }
    }
    const unsigned short* arow = &sM[wave][lm][0];

    // ---- matmul 1: [16,160] @ [160,64] ----
    v8f acc[4];
#pragma unroll
    for (int t = 0; t < 4; ++t)
#pragma unroll
      for (int q = 0; q < 8; ++q) acc[t][q] = 0.f;
#pragma unroll
    for (int kb = 0; kb < MSG_K; kb += 32) {
      Frag a; load_afrag(a, arow, kb, hi);
#pragma unroll
      for (int t = 0; t < 4; ++t) {
        Frag b; load_bfrag(b, &sW1[16 * t + lm][0], kb, hi);
        acc[t] = wmma_bf16(a, b, acc[t]);
      }
    }
    // bias + relu, re-layout C->A through LDS (cols 0..63 of this wave's tile)
#pragma unroll
    for (int t = 0; t < 4; ++t) {
      float b1 = sB1[16 * t + lm];
#pragma unroll
      for (int q = 0; q < 8; ++q) {
        float v = fmaxf(acc[t][q] + b1, 0.f);
        sM[wave][q + 8 * hi][16 * t + lm] = f2bf(v);
      }
    }
    // ---- matmul 2: [16,64] @ [64,64] ----
    v8f acc2[4];
#pragma unroll
    for (int t = 0; t < 4; ++t)
#pragma unroll
      for (int q = 0; q < 8; ++q) acc2[t][q] = 0.f;
#pragma unroll
    for (int kb = 0; kb < DD; kb += 32) {
      Frag a; load_afrag(a, arow, kb, hi);
#pragma unroll
      for (int t = 0; t < 4; ++t) {
        Frag b; load_bfrag(b, &sW2[16 * t + lm][0], kb, hi);
        acc2[t] = wmma_bf16(a, b, acc2[t]);
      }
    }
    // ---- bias + relu, scatter-add aggr, equivariant position channel ----
    float part[8];
#pragma unroll
    for (int q = 0; q < 8; ++q) part[q] = 0.f;
#pragma unroll
    for (int t = 0; t < 4; ++t) {
      float b2 = sB2[16 * t + lm];
      float wp = sWp[16 * t + lm];
#pragma unroll
      for (int q = 0; q < 8; ++q) {
        float v = fmaxf(acc2[t][q] + b2, 0.f);
        int row = q + 8 * hi;
        int di = sDst[wave][row];
        if (di >= 0) atomicAdd(&aggr[(long)di * DD + 16 * t + lm], v);
        part[q] += v * wp;
      }
    }
#pragma unroll
    for (int m = 1; m < 16; m <<= 1)
#pragma unroll
      for (int q = 0; q < 8; ++q) part[q] += __shfl_xor(part[q], m, 32);
    if (lm == 0) {
#pragma unroll
      for (int q = 0; q < 8; ++q) {
        int row = q + 8 * hi;
        int di = sDst[wave][row];
        if (di >= 0) {
          float p = part[q] + bpv;
          atomicAdd(&pos_aggr[(long)di * 3 + 0], sRel[wave][row][0] * p);
          atomicAdd(&pos_aggr[(long)di * 3 + 1], sRel[wave][row][1] * p);
          atomicAdd(&pos_aggr[(long)di * 3 + 2], sRel[wave][row][2] * p);
        }
      }
    }
  }
}

// ---------------------------------------------------------------------------
// Node kernel: h += relu(relu([h|aggr] @ Wu1 + bu1) @ Wu2 + bu2);
//              pos += pos_aggr / max(deg,1)
__global__ void __launch_bounds__(128)
k_node(float* __restrict__ h, float* __restrict__ pos,
       const float* __restrict__ aggr, const float* __restrict__ pos_a,
       const float* __restrict__ deg,
       const float* __restrict__ Wu1, const float* __restrict__ bu1,
       const float* __restrict__ Wu2, const float* __restrict__ bu2, int N) {
  __shared__ unsigned short sW1[DD][U_STRIDE];
  __shared__ unsigned short sW2[DD][W2_STRIDE];
  __shared__ float sB1[DD], sB2[DD];
  __shared__ unsigned short sU[NWAVES][16][U_STRIDE];

  const int tid = threadIdx.x;
  const int wave = tid >> 5, lane = tid & 31;
  const int hi = lane >> 4, lm = lane & 15;

  for (int idx = tid; idx < U_K * DD; idx += blockDim.x) {
    int k = idx >> 6, n = idx & 63;
    sW1[n][k] = f2bf(Wu1[k * DD + n]);
  }
  for (int idx = tid; idx < DD * DD; idx += blockDim.x) {
    int k = idx >> 6, n = idx & 63;
    sW2[n][k] = f2bf(Wu2[k * DD + n]);
  }
  if (tid < DD) { sB1[tid] = bu1[tid]; sB2[tid] = bu2[tid]; }
  __syncthreads();

  const int nTiles = (N + 15) >> 4;
  for (int tile = blockIdx.x * NWAVES + wave; tile < nTiles;
       tile += gridDim.x * NWAVES) {
    const int base = tile << 4;
    { // stage u_in = [h | aggr]
      int r = lane >> 1, half = lane & 1;
      int node = base + r;
      int nc = (node < N) ? node : (N - 1);
      unsigned short* urow = &sU[wave][r][0];
      const float4* hh = (const float4*)(h + (long)nc * DD) + half * 8;
      const float4* aa = (const float4*)(aggr + (long)nc * DD) + half * 8;
#pragma unroll
      for (int c = 0; c < 8; ++c) {
        float4 f = hh[c];
        unsigned p0 = f2bf(f.x) | ((unsigned)f2bf(f.y) << 16);
        unsigned p1 = f2bf(f.z) | ((unsigned)f2bf(f.w) << 16);
        *(uint2*)(urow + half * 32 + 4 * c) = make_uint2(p0, p1);
        f = aa[c];
        p0 = f2bf(f.x) | ((unsigned)f2bf(f.y) << 16);
        p1 = f2bf(f.z) | ((unsigned)f2bf(f.w) << 16);
        *(uint2*)(urow + 64 + half * 32 + 4 * c) = make_uint2(p0, p1);
      }
    }
    const unsigned short* arow = &sU[wave][lm][0];

    v8f acc[4];
#pragma unroll
    for (int t = 0; t < 4; ++t)
#pragma unroll
      for (int q = 0; q < 8; ++q) acc[t][q] = 0.f;
#pragma unroll
    for (int kb = 0; kb < U_K; kb += 32) {
      Frag a; load_afrag(a, arow, kb, hi);
#pragma unroll
      for (int t = 0; t < 4; ++t) {
        Frag b; load_bfrag(b, &sW1[16 * t + lm][0], kb, hi);
        acc[t] = wmma_bf16(a, b, acc[t]);
      }
    }
#pragma unroll
    for (int t = 0; t < 4; ++t) {
      float b1 = sB1[16 * t + lm];
#pragma unroll
      for (int q = 0; q < 8; ++q) {
        float v = fmaxf(acc[t][q] + b1, 0.f);
        sU[wave][q + 8 * hi][16 * t + lm] = f2bf(v);
      }
    }
    v8f acc2[4];
#pragma unroll
    for (int t = 0; t < 4; ++t)
#pragma unroll
      for (int q = 0; q < 8; ++q) acc2[t][q] = 0.f;
#pragma unroll
    for (int kb = 0; kb < DD; kb += 32) {
      Frag a; load_afrag(a, arow, kb, hi);
#pragma unroll
      for (int t = 0; t < 4; ++t) {
        Frag b; load_bfrag(b, &sW2[16 * t + lm][0], kb, hi);
        acc2[t] = wmma_bf16(a, b, acc2[t]);
      }
    }
    // residual h update (exclusive ownership of these rows -> plain RMW)
#pragma unroll
    for (int t = 0; t < 4; ++t) {
      float b2 = sB2[16 * t + lm];
#pragma unroll
      for (int q = 0; q < 8; ++q) {
        int node = base + q + 8 * hi;
        if (node < N) {
          float v = fmaxf(acc2[t][q] + b2, 0.f);
          long off = (long)node * DD + 16 * t + lm;
          h[off] += v;
        }
      }
    }
    // coordinate update: pos += pos_aggr / max(deg,1)
    for (int idx = lane; idx < 48; idx += 32) {
      int r = idx / 3, c = idx % 3;
      int node = base + r;
      if (node < N) {
        float dg = fmaxf(deg[node], 1.f);
        pos[(long)node * 3 + c] += pos_a[(long)node * 3 + c] / dg;
      }
    }
  }
}

// ---------------------------------------------------------------------------
static inline size_t align_up(size_t x) { return (x + 255) & ~(size_t)255; }

extern "C" void kernel_launch(void* const* d_in, const int* in_sizes, int n_in,
                              void* d_out, int out_size, void* d_ws, size_t ws_size,
                              hipStream_t stream) {
  (void)n_in; (void)out_size; (void)ws_size;
  const float* x     = (const float*)d_in[0];
  const float* pos   = (const float*)d_in[1];
  const float* ea    = (const float*)d_in[2];
  const int*   ei    = (const int*)d_in[3];
  const float* W_in  = (const float*)d_in[4];
  const float* b_in  = (const float*)d_in[5];
  const float* Wm1   = (const float*)d_in[6];
  const float* bm1   = (const float*)d_in[7];
  const float* Wm2   = (const float*)d_in[8];
  const float* bm2   = (const float*)d_in[9];
  const float* Wp    = (const float*)d_in[10];
  const float* bp    = (const float*)d_in[11];
  const float* Wu1   = (const float*)d_in[12];
  const float* bu1   = (const float*)d_in[13];
  const float* Wu2   = (const float*)d_in[14];
  const float* bu2   = (const float*)d_in[15];
  const float* W_out = (const float*)d_in[16];
  const float* b_out = (const float*)d_in[17];

  const int N = in_sizes[0] / 11;
  const int E = in_sizes[3] / 2;
  const int L = in_sizes[7] / DD;                 // bm1: [L,64]
  const int MSG_IN = in_sizes[6] / (L * DD);      // 2*D + EDGE_DIM + 1 = 133

  char* p = (char*)d_ws;
  float* h     = (float*)p; p += align_up((size_t)N * DD * 4);
  float* aggr  = (float*)p; p += align_up((size_t)N * DD * 4);
  float* pos_c = (float*)p; p += align_up((size_t)N * 3 * 4);
  float* pos_a = (float*)p; p += align_up((size_t)N * 3 * 4);
  float* deg   = (float*)p; p += align_up((size_t)N * 4);

  const int* src = ei;
  const int* dst = ei + E;

  k_embed<<<(int)(((long)N * DD + 255) / 256), 256, 0, stream>>>(x, W_in, b_in, h, N);
  k_prep<<<(N * 3 + 255) / 256, 256, 0, stream>>>(pos, pos_c, deg, N);
  k_deg<<<(E + 255) / 256, 256, 0, stream>>>(dst, deg, E);

  for (int l = 0; l < L; ++l) {
    k_zero<<<(int)(((long)N * DD + 255) / 256), 256, 0, stream>>>(aggr, pos_a, N);
    k_edge<<<640, 128, 0, stream>>>(h, pos_c, ea, src, dst,
        Wm1 + (size_t)l * MSG_IN * DD, bm1 + (size_t)l * DD,
        Wm2 + (size_t)l * DD * DD,     bm2 + (size_t)l * DD,
        Wp  + (size_t)l * DD,          bp  + l,
        aggr, pos_a, E, MSG_IN);
    k_node<<<320, 128, 0, stream>>>(h, pos_c, aggr, pos_a, deg,
        Wu1 + (size_t)l * 2 * DD * DD, bu1 + (size_t)l * DD,
        Wu2 + (size_t)l * DD * DD,     bu2 + (size_t)l * DD, N);
  }
  k_out<<<(N + 255) / 256, 256, 0, stream>>>(h, W_out, b_out, (float*)d_out, N);
}
